// Local_Point_Trans_5454608466700
// MI455X (gfx1250) — matrix operand order repaired and checked by reading, not compiled, not run on hardware
//
#include <hip/hip_runtime.h>

// ---------------------------------------------------------------------------
// Problem constants (fixed by the reference)
// ---------------------------------------------------------------------------
#define NPTS   8192
#define CIN    256
#define COUT   256
#define KNN    16
#define NK     (NPTS * KNN)      // 131072
#define HID    64
#define EPSBN  1e-5f
#define SLOPE  0.01f

typedef __attribute__((ext_vector_type(16))) __bf16        v16bf;
typedef __attribute__((ext_vector_type(8)))  float         v8f;
typedef __attribute__((ext_vector_type(4)))  unsigned int  v4u;

union FragBF { v16bf v; v4u q[2]; unsigned short u[16]; };
union U4U16  { v4u q; unsigned short s[8]; };

__device__ __forceinline__ float bf2f(unsigned short u) {
    unsigned int x = ((unsigned int)u) << 16;
    return __builtin_bit_cast(float, x);
}
__device__ __forceinline__ unsigned short f2bf(float f) {
    unsigned int x = __builtin_bit_cast(unsigned int, f);
    unsigned int r = x + 0x7fffu + ((x >> 16) & 1u);   // round-to-nearest-even
    return (unsigned short)(r >> 16);
}
__device__ __forceinline__ float lrelu(float x) { return x >= 0.f ? x : SLOPE * x; }

// ---------------------------------------------------------------------------
// Utility kernels
// ---------------------------------------------------------------------------
__global__ void zero_f32_kernel(float* p, int n) {
    int i = blockIdx.x * blockDim.x + threadIdx.x;
    if (i < n) p[i] = 0.f;
}

__global__ void cvt_f32_bf16_kernel(const float* __restrict__ s,
                                    unsigned short* __restrict__ d, long n) {
    long stride = (long)gridDim.x * blockDim.x;
    for (long i = blockIdx.x * (long)blockDim.x + threadIdx.x; i < n; i += stride)
        d[i] = f2bf(s[i]);
}

// wp1 is [4,64]; pad K-dim to 32 with zeros -> [32,64] bf16
__global__ void pad_wp1_kernel(const float* __restrict__ wp1,
                               unsigned short* __restrict__ dst) {
    int i = blockIdx.x * blockDim.x + threadIdx.x;   // 32*64 = 2048
    if (i >= 32 * 64) return;
    int r = i >> 6, c = i & 63;
    dst[i] = (r < 4) ? f2bf(wp1[r * 64 + c]) : (unsigned short)0;
}

// ---------------------------------------------------------------------------
// kNN: one thread per query; register-resident sorted top-16 (bubble insert)
// ---------------------------------------------------------------------------
__global__ void knn_kernel(const float* __restrict__ xyz_i,
                           const float* __restrict__ xyz_last,
                           int* __restrict__ idx) {
    int n = blockIdx.x * blockDim.x + threadIdx.x;
    if (n >= NPTS) return;
    float qx = xyz_last[n * 3 + 0];
    float qy = xyz_last[n * 3 + 1];
    float qz = xyz_last[n * 3 + 2];
    float bd[KNN]; int bi[KNN];
#pragma unroll
    for (int p = 0; p < KNN; ++p) { bd[p] = 3.4e38f; bi[p] = 0; }
    for (int j = 0; j < NPTS; ++j) {
        float dx = xyz_i[j * 3 + 0] - qx;
        float dy = xyz_i[j * 3 + 1] - qy;
        float dz = xyz_i[j * 3 + 2] - qz;
        float d = dx * dx + dy * dy + dz * dz;
        if (d < bd[KNN - 1]) {
            bd[KNN - 1] = d; bi[KNN - 1] = j;
#pragma unroll
            for (int p = KNN - 1; p > 0; --p) {
                if (bd[p] < bd[p - 1]) {
                    float td = bd[p]; bd[p] = bd[p - 1]; bd[p - 1] = td;
                    int   ti = bi[p]; bi[p] = bi[p - 1]; bi[p - 1] = ti;
                }
            }
        }
    }
#pragma unroll
    for (int p = 0; p < KNN; ++p) idx[n * KNN + p] = bi[p];
}

// ---------------------------------------------------------------------------
// bf16 WMMA GEMM: C[M,Nd] = A[M,KD] @ B[KD,Nd] + bias
// Block: 256 thr = 8 waves; 128x64 block tile; each wave owns 32x32 (2x2 WMMA
// tiles). Whole B panel [KD,64] staged transposed in LDS ONCE; A tile double-
// buffered and prefetched with global_load_async_to_lds_b128 (ASYNCcnt DMA)
// one step ahead of the WMMA consumer. K fully unrolled (KD template param).
// ---------------------------------------------------------------------------
#define BM 128
#define BN 64
#define BK 32
#define LPAD 40      // lA row pitch in ushorts: 80B, 16B-aligned rows

template <int KD, int OUTBF>
__global__ __launch_bounds__(256)
void gemm_bf16_wmma(const unsigned short* __restrict__ A,
                    const unsigned short* __restrict__ B,
                    const float* __restrict__ bias,
                    void* __restrict__ Cout,
                    int Nd) {
    constexpr int BPITCH = KD + 8;           // 16B-aligned row pitch (ushorts)
    constexpr int STEPS  = KD / BK;
    __shared__ alignas(16) unsigned short lA[2][BM][LPAD];
    __shared__ alignas(16) unsigned short lBt[BN][BPITCH];

    const int tid  = threadIdx.x;
    const int wave = tid >> 5;
    const int lane = tid & 31;
    const int wr = wave & 3;            // 0..3 -> 32-row strip
    const int wc = wave >> 2;           // 0..1 -> 32-col strip
    const long rowBase = (long)blockIdx.x * BM;
    const int  colBase = blockIdx.y * BN;
    const int m    = lane & 15;
    const int half = lane >> 4;

    // ---- stage whole B panel, transposed: lBt[n][k] ----
    {
        const int br = tid >> 3, bq = tid & 7;           // chunk (row, 8-col grp)
#pragma unroll
        for (int e = 0; e < KD / 32; ++e) {              // KD*8 chunks / 256 thr
            int r = br + e * 32;
            U4U16 ub;
            ub.q = *(const v4u*)(B + (long)r * Nd + colBase + bq * 8);
#pragma unroll
            for (int j = 0; j < 8; ++j) lBt[bq * 8 + j][r] = ub.s[j];
        }
    }

    // ---- A async-copy setup: 2 x 16B chunks per thread per step ----
    const int ar = tid >> 2, aq = tid & 3;
    const int gBase0 = (int)(((rowBase + ar)      * KD + aq * 8) * 2);
    const int gBase1 = (int)(((rowBase + ar + 64) * KD + aq * 8) * 2);
    const unsigned ldsA00 = (unsigned)(size_t)&lA[0][ar][aq * 8];
    const unsigned ldsA01 = (unsigned)(size_t)&lA[0][ar + 64][aq * 8];
    const unsigned ldsA10 = (unsigned)(size_t)&lA[1][ar][aq * 8];
    const unsigned ldsA11 = (unsigned)(size_t)&lA[1][ar + 64][aq * 8];

    // prefetch step 0 into buffer 0
    asm volatile("global_load_async_to_lds_b128 %0, %1, %2"
                 :: "v"(ldsA00), "v"(gBase0), "s"(A) : "memory");
    asm volatile("global_load_async_to_lds_b128 %0, %1, %2"
                 :: "v"(ldsA01), "v"(gBase1), "s"(A) : "memory");

    v8f acc00 = {}, acc01 = {}, acc10 = {}, acc11 = {};

#pragma unroll
    for (int s = 0; s < STEPS; ++s) {
        const int k0 = s * BK;
        // prefetch next step into the other buffer
        if (s + 1 < STEPS) {
            const unsigned l0 = ((s + 1) & 1) ? ldsA10 : ldsA00;
            const unsigned l1 = ((s + 1) & 1) ? ldsA11 : ldsA01;
            const int off = (s + 1) * BK * 2;
            asm volatile("global_load_async_to_lds_b128 %0, %1, %2"
                         :: "v"(l0), "v"(gBase0 + off), "s"(A) : "memory");
            asm volatile("global_load_async_to_lds_b128 %0, %1, %2"
                         :: "v"(l1), "v"(gBase1 + off), "s"(A) : "memory");
            asm volatile("s_wait_asynccnt 0x2" ::: "memory");
        } else {
            asm volatile("s_wait_asynccnt 0x0" ::: "memory");
        }
        __syncthreads();                           // step-s A data visible

        const int buf = s & 1;
        FragBF fa0, fa1, fb0, fb1;
        {
            const v4u* pa0 = (const v4u*)&lA[buf][wr * 32 +  0 + m][half * 8];
            const v4u* pa1 = (const v4u*)&lA[buf][wr * 32 + 16 + m][half * 8];
            fa0.q[0] = pa0[0]; fa0.q[1] = pa0[2];  // +32B = +16 ushorts in K
            fa1.q[0] = pa1[0]; fa1.q[1] = pa1[2];
            const v4u* pb0 = (const v4u*)&lBt[wc * 32 +  0 + m][k0 + half * 8];
            const v4u* pb1 = (const v4u*)&lBt[wc * 32 + 16 + m][k0 + half * 8];
            fb0.q[0] = pb0[0]; fb0.q[1] = pb0[2];
            fb1.q[0] = pb1[0]; fb1.q[1] = pb1[2];
        }

        acc00 = __builtin_amdgcn_wmma_f32_16x16x32_bf16(false, fa0.v, false, fb0.v, (short)0, acc00, false, false);
        acc01 = __builtin_amdgcn_wmma_f32_16x16x32_bf16(false, fa0.v, false, fb1.v, (short)0, acc01, false, false);
        acc10 = __builtin_amdgcn_wmma_f32_16x16x32_bf16(false, fa1.v, false, fb0.v, (short)0, acc10, false, false);
        acc11 = __builtin_amdgcn_wmma_f32_16x16x32_bf16(false, fa1.v, false, fb1.v, (short)0, acc11, false, false);
        __syncthreads();                           // reads done before overwrite
    }

    // ---- epilogue: D layout -> row = rt*16 + half*8 + r, col = ct*16 + m ----
#pragma unroll
    for (int rt = 0; rt < 2; ++rt) {
#pragma unroll
        for (int ct = 0; ct < 2; ++ct) {
            const v8f* acc = rt ? (ct ? &acc11 : &acc10) : (ct ? &acc01 : &acc00);
#pragma unroll
            for (int r = 0; r < 8; ++r) {
                long row = rowBase + wr * 32 + rt * 16 + half * 8 + r;
                int  col = colBase + wc * 32 + ct * 16 + m;
                float v = (*acc)[r] + bias[col];
                if (OUTBF) ((unsigned short*)Cout)[row * Nd + col] = f2bf(v);
                else       ((float*)Cout)[row * Nd + col] = v;
            }
        }
    }
}

// ---------------------------------------------------------------------------
// BatchNorm helpers: per-channel sum/sumsq -> scale/shift, fused lrelu apply
// ---------------------------------------------------------------------------
__global__ void bn_stats_kernel(const void* __restrict__ x, int is_bf16,
                                long rows, int C, float* __restrict__ stat) {
    int c = threadIdx.x;                       // blockDim.x == C
    float s = 0.f, sq = 0.f;
    for (long r = blockIdx.x; r < rows; r += gridDim.x) {
        float v = is_bf16 ? bf2f(((const unsigned short*)x)[r * C + c])
                          : ((const float*)x)[r * C + c];
        s += v; sq += v * v;
    }
    atomicAdd(&stat[c], s);
    atomicAdd(&stat[C + c], sq);
}

__global__ void bn_finalize_kernel(const float* __restrict__ stat,
                                   const float* __restrict__ g,
                                   const float* __restrict__ b,
                                   int C, float cnt, float* __restrict__ ss) {
    int c = threadIdx.x;
    if (c >= C) return;
    float mean = stat[c] / cnt;
    float var  = stat[C + c] / cnt - mean * mean;
    float scale = g[c] * rsqrtf(var + EPSBN);
    ss[c]     = scale;
    ss[C + c] = b[c] - mean * scale;
}

__global__ void bn_lrelu_inplace_kernel(unsigned short* __restrict__ x,
                                        long total, int Cmask,
                                        const float* __restrict__ ss, int C) {
    long stride = (long)gridDim.x * blockDim.x;
    for (long i = blockIdx.x * (long)blockDim.x + threadIdx.x; i < total; i += stride) {
        int c = (int)(i & Cmask);
        float v = bf2f(x[i]) * ss[c] + ss[C + c];
        x[i] = f2bf(lrelu(v));
    }
}

// ---------------------------------------------------------------------------
// pe-MLP input: [NK,32] bf16 = (xi - xyz_last | dt | zero-pad)
// ---------------------------------------------------------------------------
__global__ void build_ape_kernel(const float* __restrict__ xyz_i,
                                 const float* __restrict__ xyz_last,
                                 const float* __restrict__ t_i,
                                 const float* __restrict__ t_last,
                                 const int* __restrict__ idx,
                                 unsigned short* __restrict__ ape) {
    int r = blockIdx.x * blockDim.x + threadIdx.x;
    if (r >= NK) return;
    int n = r >> 4;
    int j = idx[r];
    float dx = xyz_i[j * 3 + 0] - xyz_last[n * 3 + 0];
    float dy = xyz_i[j * 3 + 1] - xyz_last[n * 3 + 1];
    float dz = xyz_i[j * 3 + 2] - xyz_last[n * 3 + 2];
    float dt = t_i[0] - t_last[0];
    long base = (long)r * 32;
    ape[base + 0] = f2bf(dx);
    ape[base + 1] = f2bf(dy);
    ape[base + 2] = f2bf(dz);
    ape[base + 3] = f2bf(dt);
#pragma unroll
    for (int c = 4; c < 32; ++c) ape[base + c] = 0;
}

// w1 = q[n] - kk[idx] + pe     (all per-channel), stored bf16
__global__ void build_w1_kernel(const float* __restrict__ q,
                                const float* __restrict__ kk,
                                const int* __restrict__ idx,
                                const unsigned short* __restrict__ pe,
                                unsigned short* __restrict__ w1, long total) {
    long stride = (long)gridDim.x * blockDim.x;
    for (long i = blockIdx.x * (long)blockDim.x + threadIdx.x; i < total; i += stride) {
        long row = i >> 8;             // / COUT
        int  c   = (int)(i & 255);
        int  n   = (int)(row >> 4);
        float v = q[(long)n * COUT + c] - kk[(long)idx[row] * COUT + c] + bf2f(pe[i]);
        w1[i] = f2bf(v);
    }
}

// ---------------------------------------------------------------------------
// Fused epilogue: BN2 + LeakyReLU + softmax over K + weighted sum with
// v = vv[idx] + pe.  One block per point, one thread per channel.
// ---------------------------------------------------------------------------
__global__ __launch_bounds__(COUT)
void finalize_kernel(const float* __restrict__ w2,
                     const float* __restrict__ ss2,
                     const float* __restrict__ vv,
                     const unsigned short* __restrict__ pe,
                     const int* __restrict__ idx,
                     float* __restrict__ out) {
    int n = blockIdx.x;
    int c = threadIdx.x;
    float scale = ss2[c], shift = ss2[COUT + c];
    float e[KNN];
    float mx = -3.4e38f;
#pragma unroll
    for (int k = 0; k < KNN; ++k) {
        long row = (long)n * KNN + k;
        float x = lrelu(w2[row * COUT + c] * scale + shift);
        e[k] = x;
        mx = fmaxf(mx, x);
    }
    float s = 0.f;
#pragma unroll
    for (int k = 0; k < KNN; ++k) { e[k] = __expf(e[k] - mx); s += e[k]; }
    float inv = 1.f / s;
    float acc = 0.f;
#pragma unroll
    for (int k = 0; k < KNN; ++k) {
        long row = (long)n * KNN + k;
        float v = vv[(long)idx[row] * COUT + c] + bf2f(pe[row * COUT + c]);
        acc += e[k] * inv * v;
    }
    out[(long)n * COUT + c] = acc;
}

// ---------------------------------------------------------------------------
// Host side
// ---------------------------------------------------------------------------
static inline long lmin(long a, long b) { return a < b ? a : b; }

extern "C" void kernel_launch(void* const* d_in, const int* in_sizes, int n_in,
                              void* d_out, int out_size, void* d_ws, size_t ws_size,
                              hipStream_t stream) {
    const float* fea_i    = (const float*)d_in[0];
    const float* fea_last = (const float*)d_in[1];
    const float* xyz_i    = (const float*)d_in[2];
    const float* xyz_last = (const float*)d_in[3];
    const float* t_i      = (const float*)d_in[4];
    const float* t_last   = (const float*)d_in[5];
    const float* wp1      = (const float*)d_in[6];
    const float* bp1      = (const float*)d_in[7];
    const float* gp       = (const float*)d_in[8];
    const float* bp       = (const float*)d_in[9];
    const float* wp2      = (const float*)d_in[10];
    const float* bp2      = (const float*)d_in[11];
    const float* wq       = (const float*)d_in[12];
    const float* bq       = (const float*)d_in[13];
    const float* wk       = (const float*)d_in[14];
    const float* bk       = (const float*)d_in[15];
    const float* wv       = (const float*)d_in[16];
    const float* bv       = (const float*)d_in[17];
    const float* gw1      = (const float*)d_in[18];
    const float* bw1      = (const float*)d_in[19];
    const float* ww       = (const float*)d_in[20];
    const float* bw       = (const float*)d_in[21];
    const float* gw2      = (const float*)d_in[22];
    const float* bw2      = (const float*)d_in[23];

    // ---- workspace carve-up ----
    char* p = (char*)d_ws;
    auto alloc = [&](size_t bytes) -> void* {
        void* r = (void*)p;
        p += (bytes + 255) & ~(size_t)255;
        return r;
    };
    int*            idx      = (int*)           alloc((size_t)NK * 4);
    float*          qbuf     = (float*)         alloc((size_t)NPTS * COUT * 4);
    float*          kkbuf    = (float*)         alloc((size_t)NPTS * COUT * 4);
    float*          vvbuf    = (float*)         alloc((size_t)NPTS * COUT * 4);
    unsigned short* feaI_bf  = (unsigned short*)alloc((size_t)NPTS * CIN * 2);
    unsigned short* feaL_bf  = (unsigned short*)alloc((size_t)NPTS * CIN * 2);
    unsigned short* wq_bf    = (unsigned short*)alloc((size_t)CIN * COUT * 2);
    unsigned short* wk_bf    = (unsigned short*)alloc((size_t)CIN * COUT * 2);
    unsigned short* wv_bf    = (unsigned short*)alloc((size_t)CIN * COUT * 2);
    unsigned short* ww_bf    = (unsigned short*)alloc((size_t)COUT * COUT * 2);
    unsigned short* wp1p_bf  = (unsigned short*)alloc((size_t)32 * HID * 2);
    unsigned short* wp2_bf   = (unsigned short*)alloc((size_t)HID * COUT * 2);
    unsigned short* ape      = (unsigned short*)alloc((size_t)NK * 32 * 2);
    unsigned short* h1       = (unsigned short*)alloc((size_t)NK * HID * 2);
    unsigned short* pe       = (unsigned short*)alloc((size_t)NK * COUT * 2);
    unsigned short* w1       = (unsigned short*)alloc((size_t)NK * COUT * 2);
    float*          w2       = (float*)         alloc((size_t)NK * COUT * 4);
    float*          stat_p   = (float*)         alloc((size_t)HID * 2 * 4);
    float*          stat_1   = (float*)         alloc((size_t)COUT * 2 * 4);
    float*          stat_2   = (float*)         alloc((size_t)COUT * 2 * 4);
    float*          ss_p     = (float*)         alloc((size_t)HID * 2 * 4);
    float*          ss_1     = (float*)         alloc((size_t)COUT * 2 * 4);
    float*          ss_2     = (float*)         alloc((size_t)COUT * 2 * 4);
    (void)ws_size; (void)n_in; (void)in_sizes; (void)out_size;

    // ---- zero BN accumulators (ws is poisoned) ----
    zero_f32_kernel<<<1, 256, 0, stream>>>(stat_p, HID * 2);
    zero_f32_kernel<<<2, 256, 0, stream>>>(stat_1, COUT * 2);
    zero_f32_kernel<<<2, 256, 0, stream>>>(stat_2, COUT * 2);

    // ---- kNN ----
    knn_kernel<<<NPTS / 256, 256, 0, stream>>>(xyz_i, xyz_last, idx);

    // ---- f32 -> bf16 conversions ----
    auto cvt = [&](const float* s, unsigned short* d, long n) {
        int g = (int)lmin((n + 255) / 256, 4096);
        cvt_f32_bf16_kernel<<<g, 256, 0, stream>>>(s, d, n);
    };
    cvt(fea_i,    feaI_bf, (long)NPTS * CIN);
    cvt(fea_last, feaL_bf, (long)NPTS * CIN);
    cvt(wq, wq_bf, (long)CIN * COUT);
    cvt(wk, wk_bf, (long)CIN * COUT);
    cvt(wv, wv_bf, (long)CIN * COUT);
    cvt(ww, ww_bf, (long)COUT * COUT);
    cvt(wp2, wp2_bf, (long)HID * COUT);
    pad_wp1_kernel<<<8, 256, 0, stream>>>(wp1, wp1p_bf);

    // ---- Q/K/V projections on pre-gather features (16x fewer FLOPs) ----
    dim3 gqkv(NPTS / BM, COUT / BN);
    gemm_bf16_wmma<CIN, 0><<<gqkv, 256, 0, stream>>>(feaL_bf, wq_bf, bq, qbuf,  COUT);
    gemm_bf16_wmma<CIN, 0><<<gqkv, 256, 0, stream>>>(feaI_bf, wk_bf, bk, kkbuf, COUT);
    gemm_bf16_wmma<CIN, 0><<<gqkv, 256, 0, stream>>>(feaI_bf, wv_bf, bv, vvbuf, COUT);

    // ---- pe MLP: [NK,32pad] -> 64 -> BN+lrelu -> 256 ----
    build_ape_kernel<<<NK / 256, 256, 0, stream>>>(xyz_i, xyz_last, t_i, t_last, idx, ape);
    dim3 gpe1(NK / BM, HID / BN);
    gemm_bf16_wmma<32, 1><<<gpe1, 256, 0, stream>>>(ape, wp1p_bf, bp1, h1, HID);
    bn_stats_kernel<<<1024, HID, 0, stream>>>(h1, 1, NK, HID, stat_p);
    bn_finalize_kernel<<<1, HID, 0, stream>>>(stat_p, gp, bp, HID, (float)NK, ss_p);
    bn_lrelu_inplace_kernel<<<4096, 256, 0, stream>>>(h1, (long)NK * HID, HID - 1, ss_p, HID);
    dim3 gpe2(NK / BM, COUT / BN);
    gemm_bf16_wmma<HID, 1><<<gpe2, 256, 0, stream>>>(h1, wp2_bf, bp2, pe, COUT);

    // ---- w1 = q - k[idx] + pe ; BN1 + lrelu ----
    build_w1_kernel<<<4096, 256, 0, stream>>>(qbuf, kkbuf, idx, pe, w1, (long)NK * COUT);
    bn_stats_kernel<<<1024, COUT, 0, stream>>>(w1, 1, NK, COUT, stat_1);
    bn_finalize_kernel<<<1, COUT, 0, stream>>>(stat_1, gw1, bw1, COUT, (float)NK, ss_1);
    bn_lrelu_inplace_kernel<<<4096, 256, 0, stream>>>(w1, (long)NK * COUT, COUT - 1, ss_1, COUT);

    // ---- dominant GEMM: w1n[NK,256] @ ww[256,256] ----
    dim3 gw2g(NK / BM, COUT / BN);
    gemm_bf16_wmma<COUT, 0><<<gw2g, 256, 0, stream>>>(w1, ww_bf, bw, w2, COUT);

    // ---- BN2 stats, then fused BN2+lrelu+softmax+weighted-sum ----
    bn_stats_kernel<<<1024, COUT, 0, stream>>>(w2, 0, NK, COUT, stat_2);
    bn_finalize_kernel<<<1, COUT, 0, stream>>>(stat_2, gw2, bw2, COUT, (float)NK, ss_2);
    finalize_kernel<<<NPTS, COUT, 0, stream>>>(w2, ss_2, vvbuf, pe, idx, (float*)d_out);
}
